// SelfAttentionModel_39651138077519
// MI455X (gfx1250) — matrix operand order repaired and checked by reading, not compiled, not run on hardware
//
#include <hip/hip_runtime.h>
#include <math.h>
#include <stdint.h>

typedef __attribute__((ext_vector_type(16))) _Float16 v16h;
typedef __attribute__((ext_vector_type(8)))  _Float16 v8h;
typedef __attribute__((ext_vector_type(8)))  float    v8f;
typedef __attribute__((ext_vector_type(4)))  int      v4i;

#define B_DIM   16
#define L_DIM   1024
#define D_DIM   1024
#define NLAYERS 3
#define OUTLEN  24

// GEMM tiling: 256 threads = 8 waves (4 M-waves x 2 N-waves);
// macro tile 128(M) x 128(N); wave tile 32x64; K step 32.
#define MT  128
#define NTW 128
#define KT  32
#define KP  40   // padded LDS row stride in halfs (80B rows -> 16B aligned, conflict-free frags)

// ---- optional CDNA5 async global->LDS path (guarded; falls back to reg pipeline) ----
#if defined(__has_builtin)
#if __has_builtin(__builtin_amdgcn_global_load_async_to_lds_b128)
#define ASYNC_LDS 1
#endif
#endif
#ifndef ASYNC_LDS
#define ASYNC_LDS 0
#endif

#if ASYNC_LDS
#define NBUF 2
#else
#define NBUF 1
#endif

__device__ __forceinline__ void wait_async0() {
#if ASYNC_LDS
#if __has_builtin(__builtin_amdgcn_s_wait_asynccnt)
    __builtin_amdgcn_s_wait_asynccnt(0);
#else
    asm volatile("s_wait_asynccnt 0" ::: "memory");
#endif
#endif
}

__device__ __forceinline__ void async_b128(const _Float16* g, _Float16* l) {
#if ASYNC_LDS
    // global side: generic and AS1 share representation; go through uintptr_t.
    __attribute__((address_space(1))) v4i* gp =
        (__attribute__((address_space(1))) v4i*)(uintptr_t)g;
    // LDS side: generic LDS pointer's low 32 bits are the LDS offset (ISA flat rule).
    __attribute__((address_space(3))) v4i* lp =
        (__attribute__((address_space(3))) v4i*)(uintptr_t)(unsigned int)(uintptr_t)l;
    __builtin_amdgcn_global_load_async_to_lds_b128(gp, lp, 0, 0);
#else
    (void)g; (void)l;
#endif
}

constexpr int EPI_SCALE     = 0; // C = acc * scale            -> f32
constexpr int EPI_F32F16    = 1; // C = acc                    -> f32 + f16
constexpr int EPI_BIAS_RELU = 2; // C = relu(acc + bias[n])    -> f16
constexpr int EPI_BIAS_RES  = 3; // C = acc + bias[n] + res    -> f32

template <bool TRANSB, int EPI>
__global__ __launch_bounds__(256)
void gemm_wmma(const _Float16* __restrict__ A, const _Float16* __restrict__ Bm,
               float* __restrict__ C, _Float16* __restrict__ C16,
               const float* __restrict__ bias, const float* __restrict__ res,
               int M, int N, int K, float scale,
               long sA, long sB, long sC)
{
    __shared__ __attribute__((aligned(16))) _Float16 ldsA[NBUF][MT * KP];
    __shared__ __attribute__((aligned(16))) _Float16 ldsB[NBUF][NTW * KP];

    const int tid   = threadIdx.x;
    const int lane  = tid & 31;
    const int wave  = tid >> 5;
    const int waveM = wave & 3;      // 0..3 -> M offset *32
    const int waveN = wave >> 2;     // 0..1 -> N offset *64
    const int bM    = blockIdx.y * MT;
    const int bN    = blockIdx.x * NTW;
    const long batch = blockIdx.z;

    const _Float16* Ab = A  + batch * sA;
    const _Float16* Bb = Bm + batch * sB;

    const int hl = lane & 15;
    const int hi = lane >> 4;

    // staging coordinates: 8-half chunks
    const int r0  = tid >> 2;            // 0..63 (rows, chunk pair r0 / r0+64)
    const int c0  = (tid & 3) * 8;       // halfs within 32-K slab
    const int kr  = tid >> 4;            // 0..15 (NN: K rows kr / kr+16)
    const int chn = tid & 15;            // NN: 8-col chunk

    v8f acc[2][4] = {};
    int cur = 0;

#if ASYNC_LDS
    // ---- prologue: async tile 0 into buf 0; NN keeps B in regs for scatter ----
    async_b128(Ab + (long)(bM +      r0) * K + c0, &ldsA[0][r0 * KP + c0]);
    async_b128(Ab + (long)(bM + 64 + r0) * K + c0, &ldsA[0][(64 + r0) * KP + c0]);
    v8h rB0, rB1;
    if (!TRANSB) {
        async_b128(Bb + (long)(bN +      r0) * K + c0, &ldsB[0][r0 * KP + c0]);
        async_b128(Bb + (long)(bN + 64 + r0) * K + c0, &ldsB[0][(64 + r0) * KP + c0]);
    } else {
        rB0 = *(const v8h*)(Bb + (long)(kr) * N + bN + chn * 8);
        rB1 = *(const v8h*)(Bb + (long)(16 + kr) * N + bN + chn * 8);
    }
#else
    // ---- prologue: sync register prefetch of tile 0 ----
    v8h rA0, rA1, rB0, rB1;
    rA0 = *(const v8h*)(Ab + (long)(bM +      r0) * K + c0);
    rA1 = *(const v8h*)(Ab + (long)(bM + 64 + r0) * K + c0);
    if (!TRANSB) {
        rB0 = *(const v8h*)(Bb + (long)(bN +      r0) * K + c0);
        rB1 = *(const v8h*)(Bb + (long)(bN + 64 + r0) * K + c0);
    } else {
        rB0 = *(const v8h*)(Bb + (long)(kr) * N + bN + chn * 8);
        rB1 = *(const v8h*)(Bb + (long)(16 + kr) * N + bN + chn * 8);
    }
#endif

    for (int k0 = 0; k0 < K; k0 += KT) {
        const int kk = k0 + KT;
#if ASYNC_LDS
        if (TRANSB) {
            // scatter this tile's B regs (transposed) into the buffer async-A is filling
            #pragma unroll
            for (int i = 0; i < 8; ++i) ldsB[cur][(chn * 8 + i) * KP + kr]        = rB0[i];
            #pragma unroll
            for (int i = 0; i < 8; ++i) ldsB[cur][(chn * 8 + i) * KP + (16 + kr)] = rB1[i];
        }
        wait_async0();
        __syncthreads();                               // buf[cur] ready; prev buf free
        const int nxt = cur ^ 1;
        if (kk < K) {
            async_b128(Ab + (long)(bM +      r0) * K + kk + c0, &ldsA[nxt][r0 * KP + c0]);
            async_b128(Ab + (long)(bM + 64 + r0) * K + kk + c0, &ldsA[nxt][(64 + r0) * KP + c0]);
            if (!TRANSB) {
                async_b128(Bb + (long)(bN +      r0) * K + kk + c0, &ldsB[nxt][r0 * KP + c0]);
                async_b128(Bb + (long)(bN + 64 + r0) * K + kk + c0, &ldsB[nxt][(64 + r0) * KP + c0]);
            } else {
                rB0 = *(const v8h*)(Bb + (long)(kk + kr) * N + bN + chn * 8);
                rB1 = *(const v8h*)(Bb + (long)(kk + 16 + kr) * N + bN + chn * 8);
            }
        }
#else
        __syncthreads();                               // everyone done reading LDS
        *(v8h*)(&ldsA[0][r0 * KP + c0])        = rA0;
        *(v8h*)(&ldsA[0][(64 + r0) * KP + c0]) = rA1;
        if (!TRANSB) {
            *(v8h*)(&ldsB[0][r0 * KP + c0])        = rB0;
            *(v8h*)(&ldsB[0][(64 + r0) * KP + c0]) = rB1;
        } else {
            #pragma unroll
            for (int i = 0; i < 8; ++i) ldsB[0][(chn * 8 + i) * KP + kr]        = rB0[i];
            #pragma unroll
            for (int i = 0; i < 8; ++i) ldsB[0][(chn * 8 + i) * KP + (16 + kr)] = rB1[i];
        }
        __syncthreads();
        if (kk < K) {                                  // overlaps with compute below
            rA0 = *(const v8h*)(Ab + (long)(bM +      r0) * K + kk + c0);
            rA1 = *(const v8h*)(Ab + (long)(bM + 64 + r0) * K + kk + c0);
            if (!TRANSB) {
                rB0 = *(const v8h*)(Bb + (long)(bN +      r0) * K + kk + c0);
                rB1 = *(const v8h*)(Bb + (long)(bN + 64 + r0) * K + kk + c0);
            } else {
                rB0 = *(const v8h*)(Bb + (long)(kk + kr) * N + bN + chn * 8);
                rB1 = *(const v8h*)(Bb + (long)(kk + 16 + kr) * N + bN + chn * 8);
            }
        }
#endif
        // ---- fragments per CDNA5 16-bit WMMA layouts ----
        const _Float16* LA = &ldsA[cur][0];
        const _Float16* LB = &ldsB[cur][0];
        v16h afrag[2], bfrag[4];
        #pragma unroll
        for (int t = 0; t < 2; ++t) {
            int row = waveM * 32 + t * 16 + hl;
            // lanes 0-15: K [0..7],[16..23]; lanes 16-31: K [8..15],[24..31]
            v8h lo = *(const v8h*)(&LA[row * KP + (hi ? 8 : 0)]);
            v8h up = *(const v8h*)(&LA[row * KP + (hi ? 24 : 16)]);
            v16h f;
            #pragma unroll
            for (int i = 0; i < 8; ++i) { f[i] = lo[i]; f[i + 8] = up[i]; }
            afrag[t] = f;
        }
        #pragma unroll
        for (int t = 0; t < 4; ++t) {
            int col = waveN * 64 + t * 16 + hl;
            // lanes 0-15: K [0..7],[16..23]; lanes 16-31: K [8..15],[24..31]
            v8h lo = *(const v8h*)(&LB[col * KP + (hi ? 8 : 0)]);
            v8h up = *(const v8h*)(&LB[col * KP + (hi ? 24 : 16)]);
            v16h f;
            #pragma unroll
            for (int i = 0; i < 8; ++i) { f[i] = lo[i]; f[i + 8] = up[i]; }
            bfrag[t] = f;
        }
        #pragma unroll
        for (int ti = 0; ti < 2; ++ti)
            #pragma unroll
            for (int tj = 0; tj < 4; ++tj)
                acc[ti][tj] = __builtin_amdgcn_wmma_f32_16x16x32_f16(
                    false, afrag[ti], false, bfrag[tj],
                    (short)0, acc[ti][tj], false, false);
#if ASYNC_LDS
        cur ^= 1;
#endif
    }

    // ---- epilogue: C VGPR j -> m = base + j + hi*8, n = base + hl ----
    #pragma unroll
    for (int ti = 0; ti < 2; ++ti) {
        #pragma unroll
        for (int tj = 0; tj < 4; ++tj) {
            int n = bN + waveN * 64 + tj * 16 + hl;
            #pragma unroll
            for (int j = 0; j < 8; ++j) {
                int m = bM + waveM * 32 + ti * 16 + j + hi * 8;
                long idx = batch * sC + (long)m * N + n;
                float v = acc[ti][tj][j];
                if (EPI == EPI_SCALE) {
                    C[idx] = v * scale;
                } else if (EPI == EPI_F32F16) {
                    C[idx] = v;
                    C16[idx] = (_Float16)v;
                } else if (EPI == EPI_BIAS_RELU) {
                    v += bias[n];
                    v = v > 0.0f ? v : 0.0f;
                    C16[idx] = (_Float16)v;
                } else { // EPI_BIAS_RES
                    v += bias[n] + res[idx];
                    C[idx] = v;
                }
            }
        }
    }
}

// ---------------- reductions ----------------
__device__ __forceinline__ float blockReduceSum(float v, float* sm8) {
    int lane = threadIdx.x & 31, wv = threadIdx.x >> 5;
    #pragma unroll
    for (int off = 16; off > 0; off >>= 1) v += __shfl_down(v, off, 32);
    __syncthreads();
    if (lane == 0) sm8[wv] = v;
    __syncthreads();
    if (threadIdx.x == 0) {
        float s = 0.0f;
        #pragma unroll
        for (int i = 0; i < 8; ++i) s += sm8[i];
        sm8[0] = s;
    }
    __syncthreads();
    return sm8[0];
}

__device__ __forceinline__ float blockReduceMax(float v, float* sm8) {
    int lane = threadIdx.x & 31, wv = threadIdx.x >> 5;
    #pragma unroll
    for (int off = 16; off > 0; off >>= 1) v = fmaxf(v, __shfl_down(v, off, 32));
    __syncthreads();
    if (lane == 0) sm8[wv] = v;
    __syncthreads();
    if (threadIdx.x == 0) {
        float s = sm8[0];
        #pragma unroll
        for (int i = 1; i < 8; ++i) s = fmaxf(s, sm8[i]);
        sm8[0] = s;
    }
    __syncthreads();
    return sm8[0];
}

// ---------------- elementwise / rowwise kernels ----------------
__global__ __launch_bounds__(256)
void add_pe_kernel(const float* __restrict__ x, const float* __restrict__ pe,
                   float* __restrict__ h, _Float16* __restrict__ h16)
{
    long i = (long)blockIdx.x * 256 + threadIdx.x;
    float v = x[i] + pe[i % ((long)L_DIM * D_DIM)];
    h[i] = v;
    h16[i] = (_Float16)v;
}

__global__ __launch_bounds__(256)
void f32_to_f16_kernel(const float* __restrict__ src, _Float16* __restrict__ dst)
{
    long i = (long)blockIdx.x * 256 + threadIdx.x;
    dst[i] = (_Float16)src[i];
}

__global__ __launch_bounds__(256)
void softmax_rows(float* __restrict__ S, _Float16* __restrict__ S16)
{
    __shared__ float sm[8];
    long row = blockIdx.x;
    float*    p   = S   + row * L_DIM;
    _Float16* p16 = S16 + row * L_DIM;
    float mx = -INFINITY;
    for (int i = threadIdx.x; i < L_DIM; i += 256) mx = fmaxf(mx, p[i]);
    mx = blockReduceMax(mx, sm);
    float s = 0.0f;
    for (int i = threadIdx.x; i < L_DIM; i += 256) {
        float e = __expf(p[i] - mx);
        p[i] = e;
        s += e;
    }
    s = blockReduceSum(s, sm);
    float inv = 1.0f / s;
    for (int i = threadIdx.x; i < L_DIM; i += 256) {
        float a = p[i] * inv;
        p[i] = a;
        p16[i] = (_Float16)a;
    }
}

__global__ __launch_bounds__(256)
void layernorm_rows(float* __restrict__ h, _Float16* __restrict__ h16,
                    const float* __restrict__ g, const float* __restrict__ b)
{
    __shared__ float sm[8];
    long row = blockIdx.x;
    float*    p   = h   + row * D_DIM;
    _Float16* p16 = h16 + row * D_DIM;
    float s = 0.0f;
    for (int i = threadIdx.x; i < D_DIM; i += 256) s += p[i];
    float mu = blockReduceSum(s, sm) * (1.0f / D_DIM);
    float v = 0.0f;
    for (int i = threadIdx.x; i < D_DIM; i += 256) { float d = p[i] - mu; v += d * d; }
    float var = blockReduceSum(v, sm) * (1.0f / D_DIM);
    float inv = rsqrtf(var + 1e-5f);
    for (int i = threadIdx.x; i < D_DIM; i += 256) {
        float y = (p[i] - mu) * inv * g[i] + b[i];
        p[i] = y;
        p16[i] = (_Float16)y;
    }
}

__global__ __launch_bounds__(256)
void fc_head(const float* __restrict__ h, const float* __restrict__ fcw,
             const float* __restrict__ fcb, float* __restrict__ z)
{
    __shared__ float sm[8];
    long row = blockIdx.x;
    const float* p = h + row * D_DIM;
    float s = 0.0f;
    for (int i = threadIdx.x; i < D_DIM; i += 256) s += p[i] * fcw[i];
    s = blockReduceSum(s, sm);
    if (threadIdx.x == 0) z[row] = s + fcb[0];
}

__global__ __launch_bounds__(256)
void final_head(const float* __restrict__ z, const float* __restrict__ fw,
                const float* __restrict__ fb, float* __restrict__ out)
{
    __shared__ float sm[8];
    int b = blockIdx.x / OUTLEN;
    int o = blockIdx.x % OUTLEN;
    const float* zr = z  + (long)b * L_DIM;
    const float* wr = fw + (long)o * L_DIM;
    float s = 0.0f;
    for (int i = threadIdx.x; i < L_DIM; i += 256) s += zr[i] * wr[i];
    s = blockReduceSum(s, sm);
    if (threadIdx.x == 0) out[b * OUTLEN + o] = s + fb[o];
}

// ---------------- launch ----------------
extern "C" void kernel_launch(void* const* d_in, const int* in_sizes, int n_in,
                              void* d_out, int out_size, void* d_ws, size_t ws_size,
                              hipStream_t stream)
{
    const float* x   = (const float*)d_in[0];
    const float* pe  = (const float*)d_in[1];
    const float* w1  = (const float*)d_in[2];
    const float* b1  = (const float*)d_in[3];
    const float* w2  = (const float*)d_in[4];
    const float* b2  = (const float*)d_in[5];
    const float* lng = (const float*)d_in[6];
    const float* lnb = (const float*)d_in[7];
    const float* fcw = (const float*)d_in[8];
    const float* fcb = (const float*)d_in[9];
    const float* fw  = (const float*)d_in[10];
    const float* fb  = (const float*)d_in[11];

    float* out  = (float*)d_out;
    float* attn = out + (size_t)B_DIM * OUTLEN;

    const long NE = (long)B_DIM * L_DIM * D_DIM;
    const long WE = (long)NLAYERS * D_DIM * D_DIM;

    char* ws = (char*)d_ws;
    size_t off = 0;
    auto alloc = [&](size_t bytes) -> char* {
        char* p = ws + off;
        off = (off + bytes + 255) & ~(size_t)255;
        return p;
    };
    float*    hf32  = (float*)   alloc(NE * 4);
    _Float16* hf16  = (_Float16*)alloc(NE * 2);
    float*    ctx32 = (float*)   alloc(NE * 4);
    _Float16* ctx16 = (_Float16*)alloc(NE * 2);
    _Float16* a16   = (_Float16*)alloc(NE * 2);
    _Float16* ff16  = (_Float16*)alloc(NE * 2);
    _Float16* w1h   = (_Float16*)alloc(WE * 2);
    _Float16* w2h   = (_Float16*)alloc(WE * 2);
    float*    z     = (float*)   alloc((size_t)B_DIM * L_DIM * 4);
    (void)ws_size; (void)in_sizes; (void)n_in; (void)out_size;

    const float scale = 1.0f / 32.0f;

    f32_to_f16_kernel<<<dim3(WE / 256), 256, 0, stream>>>(w1, w1h);
    f32_to_f16_kernel<<<dim3(WE / 256), 256, 0, stream>>>(w2, w2h);
    add_pe_kernel<<<dim3(NE / 256), 256, 0, stream>>>(x, pe, hf32, hf16);

    const dim3 gAttn(L_DIM / NTW, L_DIM / MT, B_DIM);          // (8,8,16)
    const dim3 gFFN (D_DIM / NTW, (B_DIM * L_DIM) / MT, 1);    // (8,128,1)

    for (int l = 0; l < NLAYERS; ++l) {
        float* attn_l = attn + (size_t)l * B_DIM * L_DIM * L_DIM;

        gemm_wmma<false, EPI_SCALE><<<gAttn, 256, 0, stream>>>(
            hf16, hf16, attn_l, nullptr, nullptr, nullptr,
            L_DIM, L_DIM, D_DIM, scale,
            (long)L_DIM * D_DIM, (long)L_DIM * D_DIM, (long)L_DIM * L_DIM);

        softmax_rows<<<dim3(B_DIM * L_DIM), 256, 0, stream>>>(attn_l, a16);

        gemm_wmma<true, EPI_F32F16><<<gAttn, 256, 0, stream>>>(
            a16, hf16, ctx32, ctx16, nullptr, nullptr,
            L_DIM, D_DIM, L_DIM, 1.0f,
            (long)L_DIM * L_DIM, (long)L_DIM * D_DIM, (long)L_DIM * D_DIM);

        gemm_wmma<false, EPI_BIAS_RELU><<<gFFN, 256, 0, stream>>>(
            ctx16, w1h + (long)l * D_DIM * D_DIM, nullptr, ff16,
            b1 + (long)l * D_DIM, nullptr,
            B_DIM * L_DIM, D_DIM, D_DIM, 1.0f, 0, 0, 0);

        gemm_wmma<false, EPI_BIAS_RES><<<gFFN, 256, 0, stream>>>(
            ff16, w2h + (long)l * D_DIM * D_DIM, hf32, nullptr,
            b2 + (long)l * D_DIM, ctx32,
            B_DIM * L_DIM, D_DIM, D_DIM, 1.0f, 0, 0, 0);

        layernorm_rows<<<dim3(B_DIM * L_DIM), 256, 0, stream>>>(
            hf32, hf16, lng + (long)l * D_DIM, lnb + (long)l * D_DIM);
    }

    fc_head<<<dim3(B_DIM * L_DIM), 256, 0, stream>>>(hf32, fcw, fcb, z);
    final_head<<<dim3(B_DIM * OUTLEN), 256, 0, stream>>>(z, fw, fb, out);
}
